// GAT_94489280549
// MI455X (gfx1250) — matrix operand order repaired and checked by reading, not compiled, not run on hardware
//
#include <hip/hip_runtime.h>
#include <hip/hip_bf16.h>

#define N_NODES 50000
#define IN_CH   512
#define OUT1C   256      // H1 * HID
#define NHEAD   4
#define NCLS    16
#define NEDGE   800000
#define ETOT    850000   // edges + self loops
#define NEG_SLOPE 0.2f

typedef __attribute__((ext_vector_type(16))) __bf16 v16bf;
typedef __attribute__((ext_vector_type(16))) float  v16f;
typedef __attribute__((ext_vector_type(8)))  float  v8f;

#ifndef __has_builtin
#define __has_builtin(x) 0
#endif

#if __has_builtin(__builtin_amdgcn_global_load_async_to_lds_b128)
#define HAVE_ASYNC_LDS 1
#else
#define HAVE_ASYNC_LDS 0
#endif

#if HAVE_ASYNC_LDS
typedef int v4i __attribute__((vector_size(16)));
typedef __attribute__((address_space(1))) v4i gv4i_t;   // global int4
typedef __attribute__((address_space(3))) v4i lv4i_t;   // LDS int4
# if __has_builtin(__builtin_amdgcn_s_wait_asynccnt)
#  define ASYNC_WAIT(n) __builtin_amdgcn_s_wait_asynccnt(n)
# else
#  define ASYNC_WAIT(n) asm volatile("s_wait_asynccnt %0" ::"i"(n) : "memory")
# endif
#else
# define ASYNC_WAIT(n) ((void)0)
#endif

// Copy 64 bytes (this thread's slice) global -> LDS, async if available.
__device__ __forceinline__ void async_copy64(const void* g, void* l) {
#if HAVE_ASYNC_LDS
    gv4i_t* gp = (gv4i_t*)g;
    lv4i_t* lp = (lv4i_t*)l;
    __builtin_amdgcn_global_load_async_to_lds_b128(gp, lp, 0,  0);
    __builtin_amdgcn_global_load_async_to_lds_b128(gp, lp, 16, 0);
    __builtin_amdgcn_global_load_async_to_lds_b128(gp, lp, 32, 0);
    __builtin_amdgcn_global_load_async_to_lds_b128(gp, lp, 48, 0);
#else
    const uint4* gs = (const uint4*)g;
    uint4*       ls = (uint4*)l;
    ls[0] = gs[0]; ls[1] = gs[1]; ls[2] = gs[2]; ls[3] = gs[3];
#endif
}

// Copy 32 bytes global -> LDS.
__device__ __forceinline__ void async_copy32(const void* g, void* l) {
#if HAVE_ASYNC_LDS
    gv4i_t* gp = (gv4i_t*)g;
    lv4i_t* lp = (lv4i_t*)l;
    __builtin_amdgcn_global_load_async_to_lds_b128(gp, lp, 0,  0);
    __builtin_amdgcn_global_load_async_to_lds_b128(gp, lp, 16, 0);
#else
    const uint4* gs = (const uint4*)g;
    uint4*       ls = (uint4*)l;
    ls[0] = gs[0]; ls[1] = gs[1];
#endif
}

__device__ __forceinline__ unsigned short f2bf(float f) {
    unsigned u = __float_as_uint(f);
    u += 0x7fffu + ((u >> 16) & 1u);   // round-to-nearest-even
    return (unsigned short)(u >> 16);
}

// Convert 16 f32 (four float4s) to a bf16 A-fragment via vector fptrunc (RNE);
// lets the backend select native v_cvt_*bf16_f32.
__device__ __forceinline__ v16bf cvt16(const float4& a, const float4& b,
                                       const float4& c, const float4& d) {
    v16f f;
    f[0]  = a.x; f[1]  = a.y; f[2]  = a.z; f[3]  = a.w;
    f[4]  = b.x; f[5]  = b.y; f[6]  = b.z; f[7]  = b.w;
    f[8]  = c.x; f[9]  = c.y; f[10] = c.z; f[11] = c.w;
    f[12] = d.x; f[13] = d.y; f[14] = d.z; f[15] = d.w;
    return __builtin_convertvector(f, v16bf);
}

__device__ __forceinline__ void edge_sd(const long long* __restrict__ ei, int e, int& s, int& d) {
    if (e < NEDGE) { s = (int)ei[e]; d = (int)ei[NEDGE + e]; }
    else           { s = d = e - NEDGE; }
}

__device__ __forceinline__ void atomicMaxF(float* addr, float v) {
    if (v >= 0.f) atomicMax((int*)addr, __float_as_int(v));
    else          atomicMin((unsigned int*)addr, __float_as_uint(v));
}

// ---------------------------------------------------------------------------
// Pack a row-major f32 weight [K, Ncols] into bf16 WMMA B fragments.
// Fragment order: [kc][t][lane][16 bf16]; lane's 16 values are
// (lanes 0-15: K = kc*32+{0..7,16..23} at col t*16+lane; lanes 16-31: K = kc*32+{8..15,24..31})
// matching the ISA 16-bit B 32x16 layout; each lane reads 32 contiguous bytes.
// ---------------------------------------------------------------------------
__global__ void pack_w_bf16(const float* __restrict__ W, unsigned short* __restrict__ dst,
                            int K, int Ncols) {
    int tid = blockIdx.x * blockDim.x + threadIdx.x;
    int total = (K / 32) * (Ncols / 16) * 32;
    if (tid >= total) return;
    int lane = tid & 31;
    int frag = tid >> 5;
    int nt = Ncols / 16;
    int t = frag % nt, kc = frag / nt;
    int n  = t * 16 + (lane & 15);
    int kb = (lane >> 4) << 3;                // 0 or 8
    unsigned short* o = dst + (size_t)tid * 16;
#pragma unroll
    for (int e = 0; e < 16; ++e) {
        int k = kc * 32 + kb + (e & 7) + ((e >> 3) << 4);
        o[e] = f2bf(W[(size_t)k * Ncols + n]);
    }
}

// ---------------------------------------------------------------------------
// GEMM1: H1[N,256] = x[N,512] @ W1.  One wave -> 16 rows x 256 cols (16 WMMA
// accumulators).  B fragments double-buffered in LDS via async global->LDS
// copies: all 8 waves of the block share each 16KB K-chunk of packed W1.
// A fragment (ISA 16-bit A 16x32): lane l holds row l%16; lanes 0-15 carry
// K = {0..7,16..23}, lanes 16-31 carry K = {8..15,24..31}.
// ---------------------------------------------------------------------------
__global__ void __launch_bounds__(256) gemm1_wmma(const float* __restrict__ x,
                                                  const unsigned short* __restrict__ w1b,
                                                  float* __restrict__ h1) {
    __shared__ __align__(32) unsigned short ldsB[2][16 * 32 * 16];  // 2 x 16KB
    int tid  = threadIdx.x;
    int wave = tid >> 5, lane = tid & 31;
    int tile = blockIdx.x * 8 + wave;
    int row0 = tile * 16;
    bool active = row0 < N_NODES;            // wave-uniform
    int  row0c  = active ? row0 : 0;
    int m  = lane & 15;
    int kb = (lane >> 4) << 3;               // 0 or 8
    const float* xrow = x + (size_t)(row0c + m) * IN_CH;

    const char* gB = (const char*)w1b + (size_t)tid * 64;  // this thread's slice
    async_copy64(gB, (char*)&ldsB[0][0] + tid * 64);       // chunk 0

    v8f acc[16];
#pragma unroll
    for (int t = 0; t < 16; ++t)
        acc[t] = (v8f){0.f, 0.f, 0.f, 0.f, 0.f, 0.f, 0.f, 0.f};

    for (int kc = 0; kc < 16; ++kc) {
        int cur = kc & 1;
        if (kc + 1 < 16) {
            async_copy64(gB + (size_t)(kc + 1) * 16384,
                         (char*)&ldsB[cur ^ 1][0] + tid * 64);
            ASYNC_WAIT(4);                   // chunk kc complete, kc+1 in flight
        } else {
            ASYNC_WAIT(0);
        }
        __syncthreads();                     // chunk kc visible to all waves

        __builtin_prefetch(xrow + (kc + 1) * 32, 0, 0);    // global_prefetch_b8
        const float4* p = (const float4*)(xrow + kc * 32 + kb);
        const float4* q = (const float4*)(xrow + kc * 32 + 16 + kb);
        float4 a0 = p[0], a1 = p[1], a2 = q[0], a3 = q[1];
        v16bf A = cvt16(a0, a1, a2, a3);

        const unsigned short* bsrc = &ldsB[cur][lane * 16];
#pragma unroll
        for (int t = 0; t < 16; ++t) {
            v16bf B = *(const v16bf*)(bsrc + t * 32 * 16); // ds_load_b128 x2
            acc[t] = __builtin_amdgcn_wmma_f32_16x16x32_bf16(
                false, A, false, B, (short)0, acc[t], false, false);
        }
        __syncthreads();                     // done reading before buf reuse
    }

    if (active) {
        // C/D layout: VGPR r, lane l -> row = r + 8*(l>>4), col = l&15
        int rbase = row0 + ((lane >> 4) << 3);
        int c = lane & 15;
#pragma unroll
        for (int t = 0; t < 16; ++t)
#pragma unroll
            for (int r = 0; r < 8; ++r)
                h1[(size_t)(rbase + r) * OUT1C + t * 16 + c] = acc[t][r];
    }
}

// Per-node attention logits, layer 1: one wave per node, 8 channels/lane,
// reduce over the 8 lanes of each head via shfl_xor.
__global__ void a1_logits(const float* __restrict__ h1,
                          const float* __restrict__ as_w, const float* __restrict__ ad_w,
                          float* __restrict__ a_src, float* __restrict__ a_dst) {
    int gid = blockIdx.x * blockDim.x + threadIdx.x;
    int node = gid >> 5, lane = gid & 31;
    if (node >= N_NODES) return;                 // wave-uniform
    const float4* hv = (const float4*)(h1 + (size_t)node * OUT1C + lane * 8);
    float4 v0 = hv[0], v1 = hv[1];
    const float4* ws = (const float4*)(as_w + lane * 8);
    const float4* wd = (const float4*)(ad_w + lane * 8);
    float4 s0 = ws[0], s1 = ws[1], d0 = wd[0], d1 = wd[1];
    float s = v0.x*s0.x + v0.y*s0.y + v0.z*s0.z + v0.w*s0.w
            + v1.x*s1.x + v1.y*s1.y + v1.z*s1.z + v1.w*s1.w;
    float d = v0.x*d0.x + v0.y*d0.y + v0.z*d0.z + v0.w*d0.w
            + v1.x*d1.x + v1.y*d1.y + v1.z*d1.z + v1.w*d1.w;
#pragma unroll
    for (int off = 1; off < 8; off <<= 1) {
        s += __shfl_xor(s, off, 32);
        d += __shfl_xor(d, off, 32);
    }
    if ((lane & 7) == 0) {
        a_src[node * NHEAD + (lane >> 3)] = s;
        a_dst[node * NHEAD + (lane >> 3)] = d;
    }
}

__global__ void init_l1(float* __restrict__ out1, float* __restrict__ m1, float* __restrict__ d1) {
    int i = blockIdx.x * blockDim.x + threadIdx.x;
    if (i < N_NODES * OUT1C) out1[i] = 0.f;
    if (i < N_NODES * NHEAD) { m1[i] = -3.402823466e38f; d1[i] = 0.f; }
}

__global__ void edge_max1(const long long* __restrict__ ei, const float* __restrict__ a_src,
                          const float* __restrict__ a_dst, float* __restrict__ e1,
                          float* __restrict__ m1) {
    int tid = blockIdx.x * blockDim.x + threadIdx.x;
    if (tid >= ETOT * NHEAD) return;
    int e = tid >> 2, h = tid & 3;
    int s, d; edge_sd(ei, e, s, d);
    float v = a_src[s * NHEAD + h] + a_dst[d * NHEAD + h];
    v = v >= 0.f ? v : NEG_SLOPE * v;
    e1[tid] = v;
    atomicMaxF(&m1[d * NHEAD + h], v);
}

__global__ void edge_exp1(const long long* __restrict__ ei, float* __restrict__ e1,
                          const float* __restrict__ m1, float* __restrict__ d1) {
    int tid = blockIdx.x * blockDim.x + threadIdx.x;
    if (tid >= ETOT * NHEAD) return;
    int e = tid >> 2, h = tid & 3;
    int s, d; edge_sd(ei, e, s, d);
    float ex = expf(e1[tid] - m1[d * NHEAD + h]);
    e1[tid] = ex;
    atomicAdd(&d1[d * NHEAD + h], ex);
}

// Heavy pass: 64 lanes per edge, float4 gathers from L2-resident H1,
// global_atomic_add_f32 scatter into OUT1.
__global__ void __launch_bounds__(256) edge_scatter1(const long long* __restrict__ ei,
                                                     const float* __restrict__ e1,
                                                     const float* __restrict__ den,
                                                     const float* __restrict__ h1,
                                                     float* __restrict__ out1) {
    int j = threadIdx.x & 63;
    int e = blockIdx.x * 4 + (threadIdx.x >> 6);
    if (e >= ETOT) return;
    int s, d; edge_sd(ei, e, s, d);
    int h = j >> 4;
    float alpha = e1[e * NHEAD + h] / den[d * NHEAD + h];
    float4 hv = *(const float4*)(h1 + (size_t)s * OUT1C + j * 4);
    float* o = out1 + (size_t)d * OUT1C + j * 4;
    atomicAdd(o + 0, alpha * hv.x);
    atomicAdd(o + 1, alpha * hv.y);
    atomicAdd(o + 2, alpha * hv.z);
    atomicAdd(o + 3, alpha * hv.w);
}

__global__ void h2_elu(const float* __restrict__ out1, const float* __restrict__ b1,
                       unsigned short* __restrict__ h2) {
    int i = blockIdx.x * blockDim.x + threadIdx.x;
    if (i >= N_NODES * OUT1C) return;
    float v = out1[i] + b1[i & (OUT1C - 1)];
    v = v > 0.f ? v : (expf(v) - 1.f);
    h2[i] = f2bf(v);
}

// GEMM2: Z[N,16] = elu(H1out)[N,256] @ W2.  Whole packed W2 (8KB) staged in
// LDS once per block via async copies; bf16 A already in memory.
__global__ void __launch_bounds__(256) gemm2_wmma(const unsigned short* __restrict__ h2,
                                                  const unsigned short* __restrict__ w2b,
                                                  float* __restrict__ z) {
    __shared__ __align__(32) unsigned short ldsB[8 * 32 * 16];   // 8KB
    int tid  = threadIdx.x;
    int wave = tid >> 5, lane = tid & 31;
    int tile = blockIdx.x * 8 + wave;
    int row0 = tile * 16;
    bool active = row0 < N_NODES;            // wave-uniform
    int  row0c  = active ? row0 : 0;

    async_copy32((const char*)w2b + tid * 32, (char*)&ldsB[0] + tid * 32);
    ASYNC_WAIT(0);
    __syncthreads();

    int m = lane & 15, kb = (lane >> 4) << 3;
    const unsigned short* hrow = h2 + (size_t)(row0c + m) * OUT1C;
    v8f acc = (v8f){0.f, 0.f, 0.f, 0.f, 0.f, 0.f, 0.f, 0.f};
#pragma unroll
    for (int kc = 0; kc < 8; ++kc) {
        union { v16bf v; uint4 q[2]; } A;
        A.q[0] = *(const uint4*)(hrow + kc * 32 + kb);
        A.q[1] = *(const uint4*)(hrow + kc * 32 + 16 + kb);
        v16bf B = *(const v16bf*)(&ldsB[(kc * 32 + lane) * 16]);
        acc = __builtin_amdgcn_wmma_f32_16x16x32_bf16(
            false, A.v, false, B, (short)0, acc, false, false);
    }
    if (active) {
        int rbase = row0 + ((lane >> 4) << 3), c = lane & 15;
#pragma unroll
        for (int r = 0; r < 8; ++r)
            z[(size_t)(rbase + r) * NCLS + c] = acc[r];
    }
}

__global__ void a2_logits(const float* __restrict__ z, const float* __restrict__ as_w,
                          const float* __restrict__ ad_w, float* __restrict__ a_src,
                          float* __restrict__ a_dst) {
    int gid = blockIdx.x * blockDim.x + threadIdx.x;
    int node = gid >> 4, c = gid & 15;
    int nc = node < N_NODES ? node : N_NODES - 1;
    float v = z[(size_t)nc * NCLS + c];
    float s = v * as_w[c], d = v * ad_w[c];
#pragma unroll
    for (int off = 1; off < 16; off <<= 1) {
        s += __shfl_xor(s, off, 32);
        d += __shfl_xor(d, off, 32);
    }
    if (node < N_NODES && c == 0) { a_src[node] = s; a_dst[node] = d; }
}

__global__ void init_l2(float* __restrict__ out, const float* __restrict__ b2,
                        float* __restrict__ m2, float* __restrict__ d2) {
    int i = blockIdx.x * blockDim.x + threadIdx.x;
    if (i < N_NODES * NCLS) out[i] = b2[i & 15];
    if (i < N_NODES) { m2[i] = -3.402823466e38f; d2[i] = 0.f; }
}

__global__ void edge_max2(const long long* __restrict__ ei, const float* __restrict__ a_src,
                          const float* __restrict__ a_dst, float* __restrict__ e2,
                          float* __restrict__ m2) {
    int e = blockIdx.x * blockDim.x + threadIdx.x;
    if (e >= ETOT) return;
    int s, d; edge_sd(ei, e, s, d);
    float v = a_src[s] + a_dst[d];
    v = v >= 0.f ? v : NEG_SLOPE * v;
    e2[e] = v;
    atomicMaxF(&m2[d], v);
}

__global__ void edge_exp2(const long long* __restrict__ ei, float* __restrict__ e2,
                          const float* __restrict__ m2, float* __restrict__ d2) {
    int e = blockIdx.x * blockDim.x + threadIdx.x;
    if (e >= ETOT) return;
    int s, d; edge_sd(ei, e, s, d);
    float ex = expf(e2[e] - m2[d]);
    e2[e] = ex;
    atomicAdd(&d2[d], ex);
}

__global__ void edge_scatter2(const long long* __restrict__ ei, const float* __restrict__ e2,
                              const float* __restrict__ den, const float* __restrict__ z,
                              float* __restrict__ out) {
    int tid = blockIdx.x * blockDim.x + threadIdx.x;
    if (tid >= ETOT * NCLS) return;
    int e = tid >> 4, c = tid & 15;
    int s, d; edge_sd(ei, e, s, d);
    float alpha = e2[e] / den[d];
    atomicAdd(&out[(size_t)d * NCLS + c], alpha * z[(size_t)s * NCLS + c]);
}

extern "C" void kernel_launch(void* const* d_in, const int* in_sizes, int n_in,
                              void* d_out, int out_size, void* d_ws, size_t ws_size,
                              hipStream_t stream) {
    (void)in_sizes; (void)n_in; (void)out_size; (void)ws_size;
    const float*     x   = (const float*)d_in[0];
    const long long* ei  = (const long long*)d_in[1];
    const float*     W1  = (const float*)d_in[2];
    const float*     as1 = (const float*)d_in[3];
    const float*     ad1 = (const float*)d_in[4];
    const float*     b1  = (const float*)d_in[5];
    const float*     W2  = (const float*)d_in[6];
    const float*     as2 = (const float*)d_in[7];
    const float*     ad2 = (const float*)d_in[8];
    const float*     b2  = (const float*)d_in[9];
    float* out = (float*)d_out;

    char* base = (char*)d_ws;
    size_t cur = 0;
    auto take = [&](size_t bytes) -> char* {
        char* p = base + cur;
        cur = (cur + bytes + 255) & ~(size_t)255;
        return p;
    };
    float*          H1   = (float*)take((size_t)N_NODES * OUT1C * 4);
    float*          OUT1 = (float*)take((size_t)N_NODES * OUT1C * 4);
    float*          A1s  = (float*)take((size_t)N_NODES * NHEAD * 4);
    float*          A1d  = (float*)take((size_t)N_NODES * NHEAD * 4);
    float*          M1   = (float*)take((size_t)N_NODES * NHEAD * 4);
    float*          D1   = (float*)take((size_t)N_NODES * NHEAD * 4);
    float*          E1   = (float*)take((size_t)ETOT * NHEAD * 4);
    unsigned short* W1B  = (unsigned short*)take((size_t)IN_CH * OUT1C * 2);
    unsigned short* W2B  = (unsigned short*)take((size_t)OUT1C * NCLS * 2);

    // Layer-2 buffers overlay the H1 region (H1 is dead after edge_scatter1).
    char* l2 = (char*)H1;
    auto take2 = [&](size_t bytes) -> char* {
        char* p = l2;
        l2 += (bytes + 255) & ~(size_t)255;
        return p;
    };
    unsigned short* H2  = (unsigned short*)take2((size_t)N_NODES * OUT1C * 2);
    float*          Z   = (float*)take2((size_t)N_NODES * NCLS * 4);
    float*          A2s = (float*)take2((size_t)N_NODES * 4);
    float*          A2d = (float*)take2((size_t)N_NODES * 4);
    float*          M2  = (float*)take2((size_t)N_NODES * 4);
    float*          D2  = (float*)take2((size_t)N_NODES * 4);
    float*          E2  = (float*)take2((size_t)ETOT * 4);

    const int B = 256;
    // --- layer 1 ---
    pack_w_bf16<<<(IN_CH / 32 * OUT1C / 16 * 32 + B - 1) / B, B, 0, stream>>>(W1, W1B, IN_CH, OUT1C);
    pack_w_bf16<<<1, B, 0, stream>>>(W2, W2B, OUT1C, NCLS);
    init_l1<<<(N_NODES * OUT1C + B - 1) / B, B, 0, stream>>>(OUT1, M1, D1);
    gemm1_wmma<<<(N_NODES / 16 + 7) / 8, B, 0, stream>>>(x, W1B, H1);
    a1_logits<<<(N_NODES * 32 + B - 1) / B, B, 0, stream>>>(H1, as1, ad1, A1s, A1d);
    edge_max1<<<(ETOT * NHEAD + B - 1) / B, B, 0, stream>>>(ei, A1s, A1d, E1, M1);
    edge_exp1<<<(ETOT * NHEAD + B - 1) / B, B, 0, stream>>>(ei, E1, M1, D1);
    edge_scatter1<<<(ETOT + 3) / 4, B, 0, stream>>>(ei, E1, D1, H1, OUT1);
    // --- layer 2 ---
    h2_elu<<<(N_NODES * OUT1C + B - 1) / B, B, 0, stream>>>(OUT1, b1, H2);
    gemm2_wmma<<<(N_NODES / 16 + 7) / 8, B, 0, stream>>>(H2, W2B, Z);
    a2_logits<<<(N_NODES * 16 + B - 1) / B, B, 0, stream>>>(Z, as2, ad2, A2s, A2d);
    init_l2<<<(N_NODES * NCLS + B - 1) / B, B, 0, stream>>>(out, b2, M2, D2);
    edge_max2<<<(ETOT + B - 1) / B, B, 0, stream>>>(ei, A2s, A2d, E2, M2);
    edge_exp2<<<(ETOT + B - 1) / B, B, 0, stream>>>(ei, E2, M2, D2);
    edge_scatter2<<<(ETOT * NCLS + B - 1) / B, B, 0, stream>>>(ei, E2, D2, Z, out);
}
